// CausalMultiHeadSelfAttention_38628935860421
// MI455X (gfx1250) — matrix operand order repaired and checked by reading, not compiled, not run on hardware
//
#include <hip/hip_runtime.h>
#include <hip/hip_bf16.h>

#define B_  4
#define T_  2048
#define D_  1024
#define H_  16
#define DK_ 64

typedef __attribute__((ext_vector_type(16))) __bf16 v16bf;
typedef __attribute__((ext_vector_type(8)))  __bf16 v8bf;
typedef __attribute__((ext_vector_type(8)))  float  v8f;
typedef __attribute__((ext_vector_type(4)))  float  v4f;
typedef unsigned int u32x4 __attribute__((ext_vector_type(4)));
typedef int          i32x8 __attribute__((ext_vector_type(8)));
typedef int          i32x4 __attribute__((ext_vector_type(4)));

// ---------------- WMMA fragment helpers (contiguous-k layouts for A and B, wave32) ----------------

// A-matrix 16x32 bf16: lane L holds row M=L&15; 16 contiguous K values,
// lanes 16-31 hold K=16..31.  Source is fp32 row-major with leading dim `ld`.
static __device__ inline v16bf ldA_f32(const float* __restrict__ base, int ld, int lane) {
  const float* p = base + (size_t)(lane & 15) * ld + ((lane >> 4) << 4);
  v4f a0 = *(const v4f*)(p + 0);
  v4f a1 = *(const v4f*)(p + 4);
  v4f a2 = *(const v4f*)(p + 8);
  v4f a3 = *(const v4f*)(p + 12);
  v16bf r;
#pragma unroll
  for (int i = 0; i < 4; ++i) {
    r[i]      = (__bf16)a0[i];
    r[i + 4]  = (__bf16)a1[i];
    r[i + 8]  = (__bf16)a2[i];
    r[i + 12] = (__bf16)a3[i];
  }
  return r;
}

// Same A layout, bf16 source (global Q tile or LDS P tile).
static __device__ inline v16bf ldA_bf16(const __bf16* __restrict__ base, int ld, int lane) {
  const __bf16* p = base + (size_t)(lane & 15) * ld + ((lane >> 4) << 4);
  v8bf lo = *(const v8bf*)(p);
  v8bf hi = *(const v8bf*)(p + 8);
  v16bf r;
#pragma unroll
  for (int i = 0; i < 8; ++i) { r[i] = lo[i]; r[i + 8] = hi[i]; }
  return r;
}

// B-matrix 32x16 bf16: lane L holds column N=L&15; 16 contiguous K values,
// lanes 16-31 hold K=16..31.  Memory holds B[k][n] at base[n*ld + k].
static __device__ inline v16bf ldB_f32(const float* __restrict__ base, int ld, int lane) {
  const float* p = base + (size_t)(lane & 15) * ld + ((lane >> 4) << 4);
  v4f a0 = *(const v4f*)(p + 0);
  v4f a1 = *(const v4f*)(p + 4);
  v4f a2 = *(const v4f*)(p + 8);
  v4f a3 = *(const v4f*)(p + 12);
  v16bf r;
#pragma unroll
  for (int i = 0; i < 4; ++i) {
    r[i]      = (__bf16)a0[i];
    r[i + 4]  = (__bf16)a1[i];
    r[i + 8]  = (__bf16)a2[i];
    r[i + 12] = (__bf16)a3[i];
  }
  return r;
}

static __device__ inline v16bf ldB_bf16(const __bf16* __restrict__ base, int ld, int lane) {
  const __bf16* p = base + (size_t)(lane & 15) * ld + ((lane >> 4) << 4);
  return *(const v16bf*)(p);   // 32B-aligned by construction
}

static __device__ inline v8f wmma_bf16(v16bf a, v16bf b, v8f c) {
  return __builtin_amdgcn_wmma_f32_16x16x32_bf16(false, a, false, b, (short)0, c,
                                                 false, false);
}

// ---------------- Tensor Data Mover: 2-D tile DMA  global -> LDS  -------------------
// Builds the D# descriptor per CDNA5 ISA and issues
// TENSOR_LOAD_TO_LDS.  Tracked with TENSORcnt.  All sizes in bf16 elements.
static __device__ inline void tdm_load_2d(void* lds_dst, const __bf16* gsrc,
                                          unsigned tile_w, unsigned tile_h,
                                          unsigned tensor_w, unsigned tensor_h,
                                          unsigned stride_elems) {
  const unsigned long long ga = (unsigned long long)(size_t)gsrc;
  // Flat LDS aperture truncates to addr[31:0] == LDS byte offset (ISA §10.2).
  const unsigned la = (unsigned)(size_t)lds_dst;
  u32x4 g0 = {};
  g0[0] = 1u;                                          // count=1, user flags 0
  g0[1] = la;                                          // lds_addr (bytes)
  g0[2] = (unsigned)ga;                                // global_addr[31:0]
  g0[3] = (unsigned)((ga >> 32) & 0x01FFFFFFu) | (2u << 30);  // addr[56:32], type=2
  i32x8 g1 = {};
  g1[0] = (int)(1u << 16);                             // wg_mask=0, data_size=1 (2B)
  g1[1] = (int)((tensor_w & 0xFFFFu) << 16);           // tensor_dim0[15:0]
  g1[2] = (int)((tensor_w >> 16) | ((tensor_h & 0xFFFFu) << 16));
  g1[3] = (int)((tensor_h >> 16) | (tile_w << 16));    // tile_dim0
  g1[4] = (int)tile_h;                                 // tile_dim1 (tile_dim2=0)
  g1[5] = (int)stride_elems;                           // tensor_dim0_stride[31:0]
  g1[6] = 0;
  g1[7] = 0;
  i32x4 gz = {};
#if defined(__clang_major__) && __clang_major__ >= 23
  i32x8 gz8 = {};
  __builtin_amdgcn_tensor_load_to_lds(g0, g1, gz, gz, gz8, 0);
#else
  __builtin_amdgcn_tensor_load_to_lds(g0, g1, gz, gz, 0);
#endif
}

// ---------------- GEMM: Y[m,n] = sum_k X[m,k] * W[n,k]  (bf16 WMMA, f32 acc) ----------
// MODE 0: f32 output, plain [M, D] row-major (final projection -> d_out)
// MODE 1: bf16 output scattered to [B, H, T, DK]        (Q, K)
// MODE 2: bf16 output scattered to [B, H, DK, T] (V^T so attention B-frags are contiguous)
template <int MODE>
__global__ __launch_bounds__(128) void gemm_qkv(const float* __restrict__ X,
                                                const float* __restrict__ W,
                                                void* __restrict__ Yv) {
  const int lane = threadIdx.x & 31;
  const int wave = threadIdx.x >> 5;
  const int m0 = (blockIdx.y * 4 + wave) * 16;   // 16 token rows per wave
  const int n0 = blockIdx.x * 64;                // 64 output cols per wave

  v8f acc[4] = {};
  for (int k0 = 0; k0 < D_; k0 += 32) {
    if (k0 + 32 < D_) {                          // prefetch next k-slab (speculative)
      __builtin_prefetch(X + (size_t)m0 * D_ + k0 + 32, 0, 1);
      __builtin_prefetch(W + (size_t)n0 * D_ + k0 + 32, 0, 1);
    }
    const v16bf a = ldA_f32(X + (size_t)m0 * D_ + k0, D_, lane);
#pragma unroll
    for (int j = 0; j < 4; ++j) {
      const v16bf b = ldB_f32(W + (size_t)(n0 + j * 16) * D_ + k0, D_, lane);
      acc[j] = wmma_bf16(a, b, acc[j]);
    }
  }

  const int half = lane >> 4;
  const int nl = lane & 15;
#pragma unroll
  for (int j = 0; j < 4; ++j) {
#pragma unroll
    for (int r = 0; r < 8; ++r) {
      const int grow = m0 + r + (half << 3);     // global token row in [0, B*T)
      const int col = n0 + j * 16 + nl;          // output feature
      const float v = acc[j][r];
      if (MODE == 0) {
        ((float*)Yv)[(size_t)grow * D_ + col] = v;
      } else {
        const int bb = grow >> 11;               // / T_
        const int t = grow & (T_ - 1);
        const int h = col >> 6;                  // / DK_
        const int dk = col & (DK_ - 1);
        __bf16* Yb = (__bf16*)Yv;
        if (MODE == 1)
          Yb[((((size_t)bb * H_ + h) * T_ + t) << 6) + dk] = (__bf16)v;
        else
          Yb[(((size_t)bb * H_ + h) * DK_ + dk) * T_ + t] = (__bf16)v;
      }
    }
  }
}

// ---------------- RoPE on Q,K (bf16, [B,H,T,DK], adjacent even/odd pairs) -------------
__global__ __launch_bounds__(256) void rope_kernel(__bf16* __restrict__ Qb,
                                                   __bf16* __restrict__ Kb,
                                                   const int* __restrict__ pos) {
  const int idx = blockIdx.x * blockDim.x + threadIdx.x;  // B*H*T*(DK/2) threads
  const int p = idx & 31;                 // frequency pair index
  const int t = (idx >> 5) & (T_ - 1);
  const int h = (idx >> 16) & (H_ - 1);
  const int b = idx >> 20;
  const float position = (float)pos[b * T_ + t];
  // inv_freq = theta^(-p/32) = exp(-p * ln(10000)/32)
  const float inv_freq = __expf(-(float)p * 0.28782313662425572f);
  const float ang = position * inv_freq;
  float s, c;
  __sincosf(ang, &s, &c);
  const size_t base = ((((size_t)b * H_ + h) * T_ + t) << 6) + 2 * p;
  const float q0 = (float)Qb[base], q1 = (float)Qb[base + 1];
  Qb[base]     = (__bf16)(c * q0 - s * q1);
  Qb[base + 1] = (__bf16)(s * q0 + c * q1);
  const float k0 = (float)Kb[base], k1 = (float)Kb[base + 1];
  Kb[base]     = (__bf16)(c * k0 - s * k1);
  Kb[base + 1] = (__bf16)(s * k0 + c * k1);
}

// ---------------- Flash attention: one wave per 16-query tile per (b,h) --------------
// K/V tiles are streamed into LDS by the Tensor Data Mover, double-buffered so the
// DMA of tile j+1 overlaps softmax+WMMA on tile j (TENSORcnt pipelining).
__global__ __launch_bounds__(32) void attn_kernel(const __bf16* __restrict__ Qb,
                                                  const __bf16* __restrict__ Kb,
                                                  const __bf16* __restrict__ Vt,
                                                  float* __restrict__ Of) {
  const int lane = threadIdx.x;
  const int q0 = blockIdx.x * 16;
  const int h = blockIdx.y;
  const int b = blockIdx.z;
  const __bf16* Qh = Qb + ((size_t)b * H_ + h) * T_ * DK_;
  const __bf16* Kh = Kb + ((size_t)b * H_ + h) * T_ * DK_;
  const __bf16* Vh = Vt + ((size_t)b * H_ + h) * DK_ * T_;   // [DK, T]

  // Q fragments for this tile (DK=64 -> two K=32 chunks), loaded once.
  const v16bf aq0 = ldA_bf16(Qh + (size_t)q0 * DK_, DK_, lane);
  const v16bf aq1 = ldA_bf16(Qh + (size_t)q0 * DK_ + 32, DK_, lane);

  v8f accO[4] = {};
  float rmax[8], rsum[8];
#pragma unroll
  for (int r = 0; r < 8; ++r) { rmax[r] = -__builtin_inff(); rsum[r] = 0.f; }

  __shared__ __align__(64) __bf16 kTile[2][32 * DK_];   // 32 keys x 64 dk (4 KB each)
  __shared__ __align__(64) __bf16 vTile[2][DK_ * 32];   // 64 dk x 32 keys (4 KB each)
  __shared__ __align__(64) __bf16 pTile[16 * 32];       // P for C->A layout transpose

  const int half = lane >> 4;
  const int nl = lane & 15;
  const int nkv = q0 + 16;                 // number of keys this query tile needs

  // Kick off DMA of the first K/V tiles.
  tdm_load_2d(&kTile[0][0], Kh, DK_, 32, DK_, T_, DK_);
  tdm_load_2d(&vTile[0][0], Vh, 32, DK_, T_, DK_, T_);

  int buf = 0;
  for (int j0 = 0; j0 < nkv; j0 += 32, buf ^= 1) {
    const int jn = j0 + 32;
    if (jn < nkv) {                        // prefetch next tile pair into other buffer
      tdm_load_2d(&kTile[buf ^ 1][0], Kh + (size_t)jn * DK_, DK_, 32, DK_, T_, DK_);
      tdm_load_2d(&vTile[buf ^ 1][0], Vh + jn, 32, DK_, T_, DK_, T_);
      __builtin_amdgcn_s_wait_tensorcnt(2);   // current pair complete, next in flight
    } else {
      __builtin_amdgcn_s_wait_tensorcnt(0);
    }
    const __bf16* kT = &kTile[buf][0];
    const __bf16* vT = &vTile[buf][0];

    // S = Q K^T : two 16x16 score tiles (B-frags from LDS)
    v8f s[2];
#pragma unroll
    for (int st = 0; st < 2; ++st) {
      v8f c = {};
      c = wmma_bf16(aq0, ldB_bf16(kT + (size_t)(st * 16) * DK_, DK_, lane), c);
      c = wmma_bf16(aq1, ldB_bf16(kT + (size_t)(st * 16) * DK_ + 32, DK_, lane), c);
      s[st] = c;
    }
    // scale + causal mask (C layout: row = r + 8*half, col = nl)
#pragma unroll
    for (int r = 0; r < 8; ++r) {
      const int grow = q0 + r + (half << 3);
#pragma unroll
      for (int st = 0; st < 2; ++st) {
        const int gcol = j0 + st * 16 + nl;
        const float v = s[st][r] * 0.125f;     // DK^-0.5
        s[st][r] = (gcol > grow) ? -__builtin_inff() : v;
      }
    }
    // online softmax: row reductions across the 16 lanes of each half-wave
#pragma unroll
    for (int r = 0; r < 8; ++r) {
      float tm = fmaxf(s[0][r], s[1][r]);
#pragma unroll
      for (int m = 1; m < 16; m <<= 1) tm = fmaxf(tm, __shfl_xor(tm, m, 32));
      const float nm = fmaxf(rmax[r], tm);
      const float corr = __expf(rmax[r] - nm);
      rmax[r] = nm;
      const float p0 = __expf(s[0][r] - nm);
      const float p1 = __expf(s[1][r] - nm);
      s[0][r] = p0;
      s[1][r] = p1;
      float ps = p0 + p1;
#pragma unroll
      for (int m = 1; m < 16; m <<= 1) ps += __shfl_xor(ps, m, 32);
      rsum[r] = rsum[r] * corr + ps;
#pragma unroll
      for (int j = 0; j < 4; ++j) accO[j][r] *= corr;
    }
    // P: C layout -> LDS [m][k] -> A fragment (16x32 over the 32 keys)
#pragma unroll
    for (int r = 0; r < 8; ++r) {
      const int m = r + (half << 3);
#pragma unroll
      for (int st = 0; st < 2; ++st)
        pTile[m * 32 + st * 16 + nl] = (__bf16)s[st][r];
    }
    __syncthreads();                       // single wave: dscnt ordering only
    const v16bf ap = ldA_bf16(pTile, 32, lane);
    __syncthreads();
    // O += P V : V^T rows in LDS are contiguous over the 32 local keys
#pragma unroll
    for (int j = 0; j < 4; ++j) {
      const v16bf bv = ldB_bf16(vT + (size_t)(j * 16) * 32, 32, lane);
      accO[j] = wmma_bf16(ap, bv, accO[j]);
    }
  }
  // normalize and write O back to [B, T, D] fp32 for the final projection
#pragma unroll
  for (int j = 0; j < 4; ++j) {
#pragma unroll
    for (int r = 0; r < 8; ++r) {
      const int m = r + (half << 3);
      Of[((size_t)b * T_ + q0 + m) * D_ + h * DK_ + j * 16 + nl] =
          accO[j][r] / rsum[r];
    }
  }
}

// ---------------- host-side launch --------------------------------------------------
extern "C" void kernel_launch(void* const* d_in, const int* in_sizes, int n_in,
                              void* d_out, int out_size, void* d_ws, size_t ws_size,
                              hipStream_t stream) {
  (void)in_sizes; (void)n_in; (void)out_size; (void)ws_size;
  const float* x  = (const float*)d_in[0];
  const int* tpos = (const int*)d_in[1];
  const float* WQ = (const float*)d_in[2];
  const float* WK = (const float*)d_in[3];
  const float* WV = (const float*)d_in[4];
  const float* WO = (const float*)d_in[5];
  float* out = (float*)d_out;

  char* ws = (char*)d_ws;
  const size_t nQKV = (size_t)B_ * H_ * T_ * DK_;     // 8,388,608 elems
  __bf16* Qb = (__bf16*)(ws);
  __bf16* Kb = (__bf16*)(ws + nQKV * 2);
  __bf16* Vt = (__bf16*)(ws + nQKV * 4);
  float*  Of = (float*) (ws + nQKV * 6);              // 32 MB, total ws use = 80 MB

  const dim3 gemm_grid(D_ / 64, (B_ * T_) / 64);      // 16 x 128 blocks, 4 waves each
  gemm_qkv<1><<<gemm_grid, 128, 0, stream>>>(x, WQ, (void*)Qb);
  gemm_qkv<1><<<gemm_grid, 128, 0, stream>>>(x, WK, (void*)Kb);
  gemm_qkv<2><<<gemm_grid, 128, 0, stream>>>(x, WV, (void*)Vt);

  const int rope_threads = B_ * H_ * T_ * (DK_ / 2);  // 4,194,304
  rope_kernel<<<rope_threads / 256, 256, 0, stream>>>(Qb, Kb, tpos);

  attn_kernel<<<dim3(T_ / 16, H_, B_), 32, 0, stream>>>(Qb, Kb, Vt, Of);

  gemm_qkv<0><<<gemm_grid, 128, 0, stream>>>(Of, WO, (void*)out);
}
